// DeepSeekV2MoE_56650618635055
// MI455X (gfx1250) — matrix operand order repaired and checked by reading, not compiled, not run on hardware
//
#include <hip/hip_runtime.h>

#define DINL __device__ __forceinline__

typedef __attribute__((ext_vector_type(16))) __bf16 v16bf;
typedef __attribute__((ext_vector_type(8)))  float  v8f;

union Frag16 {
  v16bf bf;
  unsigned short us[16];
  uint4 q[2];
};

constexpr int T_TOK = 4096;   // B*S
constexpr int H_DIM = 2048;
constexpr int I_DIM = 1024;
constexpr int E_EXP = 32;
constexpr int SI_DIM = 2048;
constexpr int TOPKG = 3;
constexpr int TOPK  = 6;
constexpr float RSCALE = 1.0f;
// sum(counts)==6*T exactly; pad each expert region to 128-row (M-tile) units
constexpr int HROWS = 6 * T_TOK + E_EXP * 128;  // 28672

DINL unsigned short f32_to_bf16(float x) {
  unsigned int u = __float_as_uint(x);
  u += 0x7FFFu + ((u >> 16) & 1u);   // round-to-nearest-even
  return (unsigned short)(u >> 16);
}

// A-matrix bf16 16x32 per ISA 7.12.2: lane holds row (lane&15), K-chunks at
// (lane>>4)*8 and 16+(lane>>4)*8.  p is pre-offset by (lane>>4)*8.
DINL v16bf loadA16(const unsigned short* p) {
  Frag16 f;
  f.q[0] = *(const uint4*)(p);
  f.q[1] = *(const uint4*)(p + 16);
  return f.bf;
}
// B-matrix bf16 32x16: lane holds col (lane&15), K-chunks at
// (lane>>4)*8 and 16+(lane>>4)*8.  p is pre-offset by (lane>>4)*8.
DINL v16bf loadB16(const unsigned short* p) {
  Frag16 f;
  f.q[0] = *(const uint4*)(p);
  f.q[1] = *(const uint4*)(p + 16);
  return f.bf;
}

DINL v8f wmma_bf16(v16bf a, v16bf b, v8f c) {
  return __builtin_amdgcn_wmma_f32_16x16x32_bf16(false, a, false, b, (short)0, c,
                                                 false, false);
}

DINL float silu_f(float x) {
  // x * sigmoid(x); v_rcp_f32 instead of IEEE divide (bf16 output anyway)
  return x * __builtin_amdgcn_rcpf(1.0f + __expf(-x));
}

constexpr int PF = 256;  // prefetch distance in bf16 elements (512B, 8 k-steps)

// ---------------------------------------------------------------- conversion
__global__ __launch_bounds__(256) void k_f32_to_bf16(const float* __restrict__ src,
                                                     unsigned short* __restrict__ dst,
                                                     int n4) {
  int i = blockIdx.x * 256 + threadIdx.x;
  if (i >= n4) return;
  float4 v = ((const float4*)src)[i];
  unsigned int p0 = (unsigned)f32_to_bf16(v.x) | ((unsigned)f32_to_bf16(v.y) << 16);
  unsigned int p1 = (unsigned)f32_to_bf16(v.z) | ((unsigned)f32_to_bf16(v.w) << 16);
  ((uint2*)dst)[i] = make_uint2(p0, p1);
}

// ---------------------------------------------------------------- gating
// One wave32 per token; lane == expert.  fp32 throughout (tiny GEMM).
__global__ __launch_bounds__(128) void k_gate(const float* __restrict__ x,
                                              const float* __restrict__ gw,
                                              float* __restrict__ combine) {
  const int wave = threadIdx.x >> 5;
  const int lane = threadIdx.x & 31;
  const int t = blockIdx.x * 4 + wave;

  const float4* xr = (const float4*)(x + (size_t)t * H_DIM);
  const float4* gr = (const float4*)(gw + (size_t)lane * H_DIM);
  float acc = 0.f;
  for (int k = 0; k < H_DIM / 4; ++k) {
    float4 a = xr[k], b = gr[k];
    acc += a.x * b.x + a.y * b.y + a.z * b.z + a.w * b.w;
  }
  // softmax across the 32 experts (one per lane)
  float m = acc;
  for (int s = 16; s > 0; s >>= 1) m = fmaxf(m, __shfl_xor(m, s, 32));
  float p = __expf(acc - m);
  float sum = p;
  for (int s = 16; s > 0; s >>= 1) sum += __shfl_xor(sum, s, 32);
  float score = p / sum;

  // group max (groups of 4 consecutive experts)
  float gmax = score;
  gmax = fmaxf(gmax, __shfl_xor(gmax, 1, 32));
  gmax = fmaxf(gmax, __shfl_xor(gmax, 2, 32));
  const int g = lane >> 2;
  int grank = 0;
  for (int og = 0; og < 8; ++og) {
    float gs = __shfl(gmax, og * 4, 32);
    grank += (gs > gmax) || (gs == gmax && og < g);
  }
  float masked = (grank < TOPKG) ? score : 0.f;

  // top-6 of masked scores by rank
  int rank = 0;
  for (int j = 0; j < 32; ++j) {
    float sj = __shfl(masked, j, 32);
    rank += (sj > masked) || (sj == masked && j < lane);
  }
  bool sel = (rank < TOPK) && (masked > 0.f);
  float wsel = sel ? masked : 0.f;
  float wsum = wsel;
  for (int s = 16; s > 0; s >>= 1) wsum += __shfl_xor(wsum, s, 32);
  float wout = sel ? (masked / (wsum + 1e-20f)) * RSCALE : 0.f;
  combine[(size_t)t * E_EXP + lane] = wout;
}

// ----------------------------------------------- per-expert token compaction
__global__ __launch_bounds__(256) void k_build_lists(const float* __restrict__ combine,
                                                     int* __restrict__ tok_list,
                                                     float* __restrict__ wt_list,
                                                     int* __restrict__ counts) {
  const int e = blockIdx.x;
  __shared__ int cnt;
  if (threadIdx.x == 0) cnt = 0;
  __syncthreads();
  for (int t = threadIdx.x; t < T_TOK; t += blockDim.x) {
    float w = combine[(size_t)t * E_EXP + e];
    if (w > 0.f) {
      int pos = atomicAdd(&cnt, 1);
      tok_list[e * T_TOK + pos] = t;
      wt_list[e * T_TOK + pos] = w;
    }
  }
  __syncthreads();
  if (threadIdx.x == 0) counts[e] = cnt;
}

__global__ void k_offsets(const int* __restrict__ counts, int* __restrict__ offs) {
  if (threadIdx.x == 0) {
    int acc = 0;
    for (int e = 0; e < E_EXP; ++e) {
      offs[e] = acc;
      acc += ((counts[e] + 127) >> 7) << 7;  // pad each expert to 128-row tiles
    }
  }
}

// ---------------------------------------------- routed experts: up (w1,w3)
// grid: (I/64, T/128, E); block 256 = 8 waves.
// Block tile 128M x 64N; each wave owns 2M x 2N subtiles for BOTH w1 and w3:
// 8 WMMA per k-step fed by 12 b128 loads.
__global__ __launch_bounds__(256) void k_expert_up(
    const unsigned short* __restrict__ xbf, const unsigned short* __restrict__ w1b,
    const unsigned short* __restrict__ w3b, const int* __restrict__ tok_list,
    const int* __restrict__ counts, const int* __restrict__ offs,
    unsigned short* __restrict__ h_ws) {
  const int e = blockIdx.z, mt = blockIdx.y, nt = blockIdx.x;
  const int cnt = counts[e];
  if (mt * 128 >= cnt) return;                // wave-uniform: EXEC stays all-1
  __shared__ int tokS[128];
  const int tid = threadIdx.x;
  if (tid < 128) {
    int slot = mt * 128 + tid;
    tokS[tid] = tok_list[e * T_TOK + (slot < cnt ? slot : cnt - 1)];
  }
  __syncthreads();
  const int wave = tid >> 5, lane = tid & 31;
  const int wm0 = (wave & 3) << 1;            // 2 M-subtiles
  const int wn0 = (wave >> 2) << 1;           // 2 N-subtiles
  const int lmod = lane & 15, lhalf = lane >> 4;

  const unsigned short* aP0 =
      xbf + (size_t)tokS[(wm0 + 0) * 16 + lmod] * H_DIM + lhalf * 8;
  const unsigned short* aP1 =
      xbf + (size_t)tokS[(wm0 + 1) * 16 + lmod] * H_DIM + lhalf * 8;

  const size_t ebase = (size_t)e * I_DIM * H_DIM;
  const int bcol0 = nt * 64 + wn0 * 16 + lmod;
  const unsigned short* b1P0 = w1b + ebase + (size_t)bcol0 * H_DIM + (lhalf << 3);
  const unsigned short* b1P1 = b1P0 + (size_t)16 * H_DIM;
  const unsigned short* b3P0 = w3b + ebase + (size_t)bcol0 * H_DIM + (lhalf << 3);
  const unsigned short* b3P1 = b3P0 + (size_t)16 * H_DIM;

  v8f a1[2][2] = {}, a3[2][2] = {};
#pragma unroll 2
  for (int kb = 0; kb < H_DIM; kb += 32) {
    __builtin_prefetch((const void*)(aP0 + kb + PF), 0, 3);
    __builtin_prefetch((const void*)(b1P0 + kb + PF), 0, 3);
    __builtin_prefetch((const void*)(b3P0 + kb + PF), 0, 3);
    v16bf af0 = loadA16(aP0 + kb);
    v16bf af1 = loadA16(aP1 + kb);
    v16bf b10 = loadB16(b1P0 + kb);
    v16bf b11 = loadB16(b1P1 + kb);
    v16bf b30 = loadB16(b3P0 + kb);
    v16bf b31 = loadB16(b3P1 + kb);
    a1[0][0] = wmma_bf16(af0, b10, a1[0][0]);
    a1[0][1] = wmma_bf16(af0, b11, a1[0][1]);
    a1[1][0] = wmma_bf16(af1, b10, a1[1][0]);
    a1[1][1] = wmma_bf16(af1, b11, a1[1][1]);
    a3[0][0] = wmma_bf16(af0, b30, a3[0][0]);
    a3[0][1] = wmma_bf16(af0, b31, a3[0][1]);
    a3[1][0] = wmma_bf16(af1, b30, a3[1][0]);
    a3[1][1] = wmma_bf16(af1, b31, a3[1][1]);
  }
  const size_t rbase = (size_t)offs[e] + mt * 128;
#pragma unroll
  for (int mi = 0; mi < 2; ++mi) {
    const size_t row0 = rbase + (wm0 + mi) * 16 + lhalf * 8;
#pragma unroll
    for (int nj = 0; nj < 2; ++nj) {
      const int col = nt * 64 + (wn0 + nj) * 16 + lmod;
#pragma unroll
      for (int r = 0; r < 8; ++r) {
        float hv = silu_f(a1[mi][nj][r]) * a3[mi][nj][r];
        h_ws[(row0 + r) * I_DIM + col] = f32_to_bf16(hv);
      }
    }
  }
}

// ---------------------------------------------- routed experts: down (w2)
// grid: (H/128, T/128, E); block tile 128M x 128N; wave owns 2M x 4N.
__global__ __launch_bounds__(256) void k_expert_down(
    const unsigned short* __restrict__ h_ws, const unsigned short* __restrict__ w2b,
    const int* __restrict__ tok_list, const float* __restrict__ wt_list,
    const int* __restrict__ counts, const int* __restrict__ offs,
    float* __restrict__ out) {
  const int e = blockIdx.z, mt = blockIdx.y, nt = blockIdx.x;
  const int cnt = counts[e];
  if (mt * 128 >= cnt) return;
  __shared__ int tokS[128];
  __shared__ float wtS[128];
  const int tid = threadIdx.x;
  if (tid < 128) {
    int slot = mt * 128 + tid;
    int cs = slot < cnt ? slot : cnt - 1;
    tokS[tid] = tok_list[e * T_TOK + cs];
    wtS[tid]  = wt_list[e * T_TOK + cs];
  }
  __syncthreads();
  const int wave = tid >> 5, lane = tid & 31;
  const int wm0 = (wave & 3) << 1;            // 2 M-subtiles
  const int wn0 = (wave >> 2) << 2;           // 4 N-subtiles
  const int lmod = lane & 15, lhalf = lane >> 4;

  const size_t rbase = (size_t)offs[e] + mt * 128;
  const unsigned short* aP0 =
      h_ws + (rbase + (wm0 + 0) * 16 + lmod) * I_DIM + lhalf * 8;
  const unsigned short* aP1 =
      h_ws + (rbase + (wm0 + 1) * 16 + lmod) * I_DIM + lhalf * 8;

  const size_t ebase = (size_t)e * H_DIM * I_DIM;
  const int bcol0 = nt * 128 + wn0 * 16 + lmod;
  const unsigned short* bP0 = w2b + ebase + (size_t)bcol0 * I_DIM + (lhalf << 3);
  const unsigned short* bP1 = bP0 + (size_t)16 * I_DIM;
  const unsigned short* bP2 = bP0 + (size_t)32 * I_DIM;
  const unsigned short* bP3 = bP0 + (size_t)48 * I_DIM;

  v8f c[2][4] = {};
#pragma unroll 2
  for (int kb = 0; kb < I_DIM; kb += 32) {
    __builtin_prefetch((const void*)(aP0 + kb + PF), 0, 3);
    __builtin_prefetch((const void*)(bP0 + kb + PF), 0, 3);
    v16bf af0 = loadA16(aP0 + kb);
    v16bf af1 = loadA16(aP1 + kb);
    v16bf b0 = loadB16(bP0 + kb);
    v16bf b1 = loadB16(bP1 + kb);
    v16bf b2 = loadB16(bP2 + kb);
    v16bf b3 = loadB16(bP3 + kb);
    c[0][0] = wmma_bf16(af0, b0, c[0][0]);
    c[0][1] = wmma_bf16(af0, b1, c[0][1]);
    c[0][2] = wmma_bf16(af0, b2, c[0][2]);
    c[0][3] = wmma_bf16(af0, b3, c[0][3]);
    c[1][0] = wmma_bf16(af1, b0, c[1][0]);
    c[1][1] = wmma_bf16(af1, b1, c[1][1]);
    c[1][2] = wmma_bf16(af1, b2, c[1][2]);
    c[1][3] = wmma_bf16(af1, b3, c[1][3]);
  }
#pragma unroll
  for (int mi = 0; mi < 2; ++mi) {
    const int rloc = (wm0 + mi) * 16 + lhalf * 8;
#pragma unroll
    for (int r = 0; r < 8; ++r) {
      int slot = mt * 128 + rloc + r;
      if (slot < cnt) {
        int tok = tokS[rloc + r];
        float w = wtS[rloc + r];
        float* op = out + (size_t)tok * H_DIM;
#pragma unroll
        for (int nj = 0; nj < 4; ++nj) {
          int col = nt * 128 + (wn0 + nj) * 16 + lmod;
          atomicAdd(op + col, w * c[mi][nj][r]);
        }
      }
    }
  }
}

// ---------------------------------------------- shared experts: up
// grid: (SI/64, T/128); block tile 128M x 64N; wave owns 2M x 2N x {sw1,sw3}.
__global__ __launch_bounds__(256) void k_shared_up(
    const unsigned short* __restrict__ xbf, const unsigned short* __restrict__ sw1b,
    const unsigned short* __restrict__ sw3b, unsigned short* __restrict__ hs_ws) {
  const int mt = blockIdx.y, nt = blockIdx.x;
  const int tid = threadIdx.x;
  const int wave = tid >> 5, lane = tid & 31;
  const int wm0 = (wave & 3) << 1, wn0 = (wave >> 2) << 1;
  const int lmod = lane & 15, lhalf = lane >> 4;

  const int row0i = mt * 128 + (wm0 + 0) * 16 + lmod;
  const int row1i = mt * 128 + (wm0 + 1) * 16 + lmod;
  const unsigned short* aP0 = xbf + (size_t)row0i * H_DIM + lhalf * 8;
  const unsigned short* aP1 = xbf + (size_t)row1i * H_DIM + lhalf * 8;
  const int bcol0 = nt * 64 + wn0 * 16 + lmod;
  const unsigned short* b1P0 = sw1b + (size_t)bcol0 * H_DIM + (lhalf << 3);
  const unsigned short* b1P1 = b1P0 + (size_t)16 * H_DIM;
  const unsigned short* b3P0 = sw3b + (size_t)bcol0 * H_DIM + (lhalf << 3);
  const unsigned short* b3P1 = b3P0 + (size_t)16 * H_DIM;

  v8f a1[2][2] = {}, a3[2][2] = {};
#pragma unroll 2
  for (int kb = 0; kb < H_DIM; kb += 32) {
    __builtin_prefetch((const void*)(aP0 + kb + PF), 0, 3);
    __builtin_prefetch((const void*)(b1P0 + kb + PF), 0, 3);
    __builtin_prefetch((const void*)(b3P0 + kb + PF), 0, 3);
    v16bf af0 = loadA16(aP0 + kb);
    v16bf af1 = loadA16(aP1 + kb);
    v16bf b10 = loadB16(b1P0 + kb);
    v16bf b11 = loadB16(b1P1 + kb);
    v16bf b30 = loadB16(b3P0 + kb);
    v16bf b31 = loadB16(b3P1 + kb);
    a1[0][0] = wmma_bf16(af0, b10, a1[0][0]);
    a1[0][1] = wmma_bf16(af0, b11, a1[0][1]);
    a1[1][0] = wmma_bf16(af1, b10, a1[1][0]);
    a1[1][1] = wmma_bf16(af1, b11, a1[1][1]);
    a3[0][0] = wmma_bf16(af0, b30, a3[0][0]);
    a3[0][1] = wmma_bf16(af0, b31, a3[0][1]);
    a3[1][0] = wmma_bf16(af1, b30, a3[1][0]);
    a3[1][1] = wmma_bf16(af1, b31, a3[1][1]);
  }
#pragma unroll
  for (int mi = 0; mi < 2; ++mi) {
    const size_t row0 = (size_t)mt * 128 + (wm0 + mi) * 16 + lhalf * 8;
#pragma unroll
    for (int nj = 0; nj < 2; ++nj) {
      const int col = nt * 64 + (wn0 + nj) * 16 + lmod;
#pragma unroll
      for (int r = 0; r < 8; ++r) {
        float hv = silu_f(a1[mi][nj][r]) * a3[mi][nj][r];
        hs_ws[(row0 + r) * SI_DIM + col] = f32_to_bf16(hv);
      }
    }
  }
}

// ---------------------------------------------- shared experts: down (writes out)
// grid: (H/128, T/128); block tile 128M x 128N; wave owns 2M x 4N.
__global__ __launch_bounds__(256) void k_shared_down(
    const unsigned short* __restrict__ hs_ws, const unsigned short* __restrict__ sw2b,
    float* __restrict__ out) {
  const int mt = blockIdx.y, nt = blockIdx.x;
  const int tid = threadIdx.x;
  const int wave = tid >> 5, lane = tid & 31;
  const int wm0 = (wave & 3) << 1, wn0 = (wave >> 2) << 2;
  const int lmod = lane & 15, lhalf = lane >> 4;

  const int row0i = mt * 128 + (wm0 + 0) * 16 + lmod;
  const int row1i = mt * 128 + (wm0 + 1) * 16 + lmod;
  const unsigned short* aP0 = hs_ws + (size_t)row0i * SI_DIM + lhalf * 8;
  const unsigned short* aP1 = hs_ws + (size_t)row1i * SI_DIM + lhalf * 8;
  const int bcol0 = nt * 128 + wn0 * 16 + lmod;
  const unsigned short* bP0 = sw2b + (size_t)bcol0 * SI_DIM + (lhalf << 3);
  const unsigned short* bP1 = bP0 + (size_t)16 * SI_DIM;
  const unsigned short* bP2 = bP0 + (size_t)32 * SI_DIM;
  const unsigned short* bP3 = bP0 + (size_t)48 * SI_DIM;

  v8f c[2][4] = {};
#pragma unroll 2
  for (int kb = 0; kb < SI_DIM; kb += 32) {
    __builtin_prefetch((const void*)(aP0 + kb + PF), 0, 3);
    __builtin_prefetch((const void*)(bP0 + kb + PF), 0, 3);
    v16bf af0 = loadA16(aP0 + kb);
    v16bf af1 = loadA16(aP1 + kb);
    v16bf b0 = loadB16(bP0 + kb);
    v16bf b1 = loadB16(bP1 + kb);
    v16bf b2 = loadB16(bP2 + kb);
    v16bf b3 = loadB16(bP3 + kb);
    c[0][0] = wmma_bf16(af0, b0, c[0][0]);
    c[0][1] = wmma_bf16(af0, b1, c[0][1]);
    c[0][2] = wmma_bf16(af0, b2, c[0][2]);
    c[0][3] = wmma_bf16(af0, b3, c[0][3]);
    c[1][0] = wmma_bf16(af1, b0, c[1][0]);
    c[1][1] = wmma_bf16(af1, b1, c[1][1]);
    c[1][2] = wmma_bf16(af1, b2, c[1][2]);
    c[1][3] = wmma_bf16(af1, b3, c[1][3]);
  }
#pragma unroll
  for (int mi = 0; mi < 2; ++mi) {
    const size_t row0 = (size_t)mt * 128 + (wm0 + mi) * 16 + lhalf * 8;
#pragma unroll
    for (int nj = 0; nj < 4; ++nj) {
      const int col = nt * 128 + (wn0 + nj) * 16 + lmod;
#pragma unroll
      for (int r = 0; r < 8; ++r) {
        out[(row0 + r) * H_DIM + col] = c[mi][nj][r];
      }
    }
  }
}

// =========================================================================
extern "C" void kernel_launch(void* const* d_in, const int* in_sizes, int n_in,
                              void* d_out, int out_size, void* d_ws, size_t ws_size,
                              hipStream_t stream) {
  (void)in_sizes; (void)n_in; (void)out_size; (void)ws_size;
  const float* x   = (const float*)d_in[0];
  const float* gw  = (const float*)d_in[1];
  const float* w1  = (const float*)d_in[2];
  const float* w2  = (const float*)d_in[3];
  const float* w3  = (const float*)d_in[4];
  const float* sw1 = (const float*)d_in[5];
  const float* sw2 = (const float*)d_in[6];
  const float* sw3 = (const float*)d_in[7];
  float* out = (float*)d_out;

  // workspace carve (~520 MB: bf16 weights 428MB, h 58.7MB, hs 16.8MB, x 16.8MB)
  char* ws = (char*)d_ws;
  size_t off = 0;
  auto carve = [&](size_t bytes) -> char* {
    char* p = ws + off;
    off = (off + bytes + 255) & ~(size_t)255;
    return p;
  };
  unsigned short* xbf   = (unsigned short*)carve((size_t)T_TOK * H_DIM * 2);
  unsigned short* w1b   = (unsigned short*)carve((size_t)E_EXP * I_DIM * H_DIM * 2);
  unsigned short* w2b   = (unsigned short*)carve((size_t)E_EXP * H_DIM * I_DIM * 2);
  unsigned short* w3b   = (unsigned short*)carve((size_t)E_EXP * I_DIM * H_DIM * 2);
  unsigned short* sw1b  = (unsigned short*)carve((size_t)SI_DIM * H_DIM * 2);
  unsigned short* sw2b  = (unsigned short*)carve((size_t)H_DIM * SI_DIM * 2);
  unsigned short* sw3b  = (unsigned short*)carve((size_t)SI_DIM * H_DIM * 2);
  float* combine        = (float*)carve((size_t)T_TOK * E_EXP * 4);
  int* tok_list         = (int*)carve((size_t)E_EXP * T_TOK * 4);
  float* wt_list        = (float*)carve((size_t)E_EXP * T_TOK * 4);
  int* counts           = (int*)carve((size_t)E_EXP * 4);
  int* offs             = (int*)carve((size_t)E_EXP * 4);
  unsigned short* h_ws  = (unsigned short*)carve((size_t)HROWS * I_DIM * 2);
  unsigned short* hs_ws = (unsigned short*)carve((size_t)T_TOK * SI_DIM * 2);

  auto cvt = [&](const float* s, unsigned short* d, size_t n) {
    int n4 = (int)(n / 4);
    k_f32_to_bf16<<<dim3((n4 + 255) / 256), dim3(256), 0, stream>>>(s, d, n4);
  };
  cvt(x,   xbf,  (size_t)T_TOK * H_DIM);
  cvt(w1,  w1b,  (size_t)E_EXP * I_DIM * H_DIM);
  cvt(w2,  w2b,  (size_t)E_EXP * H_DIM * I_DIM);
  cvt(w3,  w3b,  (size_t)E_EXP * I_DIM * H_DIM);
  cvt(sw1, sw1b, (size_t)SI_DIM * H_DIM);
  cvt(sw2, sw2b, (size_t)H_DIM * SI_DIM);
  cvt(sw3, sw3b, (size_t)SI_DIM * H_DIM);

  k_gate<<<dim3(T_TOK / 4), dim3(128), 0, stream>>>(x, gw, combine);
  k_build_lists<<<dim3(E_EXP), dim3(256), 0, stream>>>(combine, tok_list, wt_list, counts);
  k_offsets<<<dim3(1), dim3(32), 0, stream>>>(counts, offs);

  k_shared_up<<<dim3(SI_DIM / 64, T_TOK / 128), dim3(256), 0, stream>>>(xbf, sw1b, sw3b, hs_ws);
  k_shared_down<<<dim3(H_DIM / 128, T_TOK / 128), dim3(256), 0, stream>>>(hs_ws, sw2b, out);

  k_expert_up<<<dim3(I_DIM / 64, T_TOK / 128, E_EXP), dim3(256), 0, stream>>>(
      xbf, w1b, w3b, tok_list, counts, offs, h_ws);
  k_expert_down<<<dim3(H_DIM / 128, T_TOK / 128, E_EXP), dim3(256), 0, stream>>>(
      h_ws, w2b, tok_list, wt_list, counts, offs, out);
}
